// S4ModuleEfficient_33775622815806
// MI455X (gfx1250) — compile-verified, matrix operand order checked by reading, not run on hardware
//
#include <hip/hip_runtime.h>

#define D_MODEL 1024
#define L_SEQ   4096
#define BATCH   2
#define N_STATE 64
#define M_TOTAL (BATCH * L_SEQ)   // 8192

typedef __attribute__((ext_vector_type(16))) __bf16 v16bf;
typedef __attribute__((ext_vector_type(8)))  float  v8f;
typedef __attribute__((ext_vector_type(4)))  float  f32x4;

__device__ __forceinline__ unsigned short f2bf_rne(float f) {
  unsigned u = __float_as_uint(f);
  u += 0x7FFFu + ((u >> 16) & 1u);
  return (unsigned short)(u >> 16);
}
__device__ __forceinline__ float bf2f(unsigned short h) {
  return __uint_as_float(((unsigned)h) << 16);
}

// ---------------------------------------------------------------------------
// GEMM: C[m,n] = sum_k A[m,k] * W[n,k] + bias[n]   (both operands k-major)
// bf16x3 split (hi*hi + hi*lo + lo*hi), f32 WMMA accumulation.
// BM=BN=128, KC=32; 8 waves (2x4), each owns a 64x32 tile = 4x2 WMMA frags.
// LDS ping-pong double buffering: stage chunk k+1 while computing chunk k.
// TRANSA=true reads A as y[b][k][t] with m=(b,t)  (t contiguous -> coalesced).
// ---------------------------------------------------------------------------
template<bool TRANSA>
__global__ __launch_bounds__(256)
void gemm_bf16x3(const float* __restrict__ A, const float* __restrict__ W,
                 const float* __restrict__ bias, float* __restrict__ Cout,
                 int N, int K) {
  // fragment-linear LDS: [buf][plane hi/lo][subtile 0..7][lane 0..31][slot 0..15]
  __shared__ __align__(32) unsigned short sA[2][2][4096];
  __shared__ __align__(32) unsigned short sB[2][2][4096];

  const int tid  = threadIdx.x;
  const int lane = tid & 31;
  const int wave = tid >> 5;
  const int wm   = wave >> 2;       // 0..1  -> 64 rows
  const int wn   = wave & 3;        // 0..3  -> 32 cols
  const int m0   = blockIdx.x * 128;
  const int n0   = blockIdx.y * 128;

  v8f acc[4][2];
  #pragma unroll
  for (int i = 0; i < 4; ++i)
    #pragma unroll
    for (int j = 0; j < 2; ++j) acc[i][j] = (v8f)(0.0f);

  auto stage = [&](int buf, int k0) {
    // ---- A tile (128 x 32) as bf16 hi/lo in fragment order ----
    #pragma unroll
    for (int it = 0; it < 16; ++it) {
      int idx = tid + it * 256;
      int mloc, kk;
      float val;
      if constexpr (TRANSA) {
        mloc = idx & 127; kk = idx >> 7;          // consecutive threads -> consecutive t
        int mg = m0 + mloc;
        int bb = mg >> 12;                        // / L_SEQ
        int tt = mg & (L_SEQ - 1);
        val = A[((size_t)bb * K + (k0 + kk)) * (size_t)L_SEQ + tt];
      } else {
        kk = idx & 31; mloc = idx >> 5;           // consecutive threads -> consecutive k
        val = A[(size_t)(m0 + mloc) * K + k0 + kk];
      }
      unsigned short hi = f2bf_rne(val);
      unsigned short lo = f2bf_rne(val - bf2f(hi));
      int sub  = mloc >> 4;
      int ln   = (mloc & 15) + 16 * ((kk >> 3) & 1);
      int slot = (kk & 7) + 8 * (kk >> 4);
      int off  = sub * 512 + ln * 16 + slot;
      sA[buf][0][off] = hi; sA[buf][1][off] = lo;
    }
    // ---- B tile: W[n0+nloc, k0+kk] ----
    #pragma unroll
    for (int it = 0; it < 16; ++it) {
      int idx  = tid + it * 256;
      int kk   = idx & 31;
      int nloc = idx >> 5;
      float val = W[(size_t)(n0 + nloc) * K + k0 + kk];
      unsigned short hi = f2bf_rne(val);
      unsigned short lo = f2bf_rne(val - bf2f(hi));
      int sub  = nloc >> 4;
      int ln   = (nloc & 15) + 16 * ((kk >> 3) & 1);
      int slot = (kk & 7) + 8 * (kk >> 4);
      int off  = sub * 512 + ln * 16 + slot;
      sB[buf][0][off] = hi; sB[buf][1][off] = lo;
    }
  };

  stage(0, 0);
  __syncthreads();

  const int KC = K / 32;
  for (int kc = 0; kc < KC; ++kc) {
    const int cur = kc & 1;
    if (kc + 1 < KC) stage(cur ^ 1, (kc + 1) * 32);  // overlap with WMMAs below

    v16bf aHi[4], aLo[4], bHi[2], bLo[2];
    #pragma unroll
    for (int i = 0; i < 4; ++i) {
      int sub = wm * 4 + i;
      aHi[i] = *reinterpret_cast<const v16bf*>(&sA[cur][0][sub * 512 + lane * 16]);
      aLo[i] = *reinterpret_cast<const v16bf*>(&sA[cur][1][sub * 512 + lane * 16]);
    }
    #pragma unroll
    for (int j = 0; j < 2; ++j) {
      int sub = wn * 2 + j;
      bHi[j] = *reinterpret_cast<const v16bf*>(&sB[cur][0][sub * 512 + lane * 16]);
      bLo[j] = *reinterpret_cast<const v16bf*>(&sB[cur][1][sub * 512 + lane * 16]);
    }
    #pragma unroll
    for (int i = 0; i < 4; ++i)
      #pragma unroll
      for (int j = 0; j < 2; ++j) {
        acc[i][j] = __builtin_amdgcn_wmma_f32_16x16x32_bf16(
            false, aHi[i], false, bHi[j], (short)0, acc[i][j], false, false);
        acc[i][j] = __builtin_amdgcn_wmma_f32_16x16x32_bf16(
            false, aHi[i], false, bLo[j], (short)0, acc[i][j], false, false);
        acc[i][j] = __builtin_amdgcn_wmma_f32_16x16x32_bf16(
            false, aLo[i], false, bHi[j], (short)0, acc[i][j], false, false);
      }
    __syncthreads();
  }

  // ---- epilogue: C/D layout -> m = 16i + v + 8*(lane/16), n = 16j + lane%16
  const int rowOff = 8 * (lane >> 4);
  const int colIn  = lane & 15;
  #pragma unroll
  for (int i = 0; i < 4; ++i) {
    int mBase = m0 + wm * 64 + i * 16 + rowOff;
    #pragma unroll
    for (int j = 0; j < 2; ++j) {
      int col  = n0 + wn * 32 + j * 16 + colIn;
      float bv = bias[col];
      #pragma unroll
      for (int v = 0; v < 8; ++v)
        Cout[(size_t)(mBase + v) * N + col] = acc[i][j][v] + bv;
    }
  }
}

// ---------------------------------------------------------------------------
// 32x32 tiled transpose: x1[(b*L + t), d] -> u[b][d][t]
// ---------------------------------------------------------------------------
__global__ __launch_bounds__(256)
void transpose_ld(const float* __restrict__ x1, float* __restrict__ u) {
  __shared__ float tile[32][33];
  const int t0 = blockIdx.x * 32;
  const int d0 = blockIdx.y * 32;
  const int b  = blockIdx.z;
  const int tx = threadIdx.x, ty = threadIdx.y;
  #pragma unroll
  for (int i = 0; i < 4; ++i)
    tile[ty + i * 8][tx] =
        x1[(size_t)(b * L_SEQ + t0 + ty + i * 8) * D_MODEL + d0 + tx];
  __syncthreads();
  #pragma unroll
  for (int i = 0; i < 4; ++i)
    u[((size_t)b * D_MODEL + d0 + ty + i * 8) * L_SEQ + t0 + tx] =
        tile[tx][ty + i * 8];
}

// ---------------------------------------------------------------------------
// SSM scan: one wave32 per (b,d); lane owns states {lane, lane+32}.
// h = dA*h + u;  y = sum_n w_n*h_n + Dp*u,  w = C*B*dt, dA = exp(-exp(-Alog)*dt)
// u chunks streamed Global->LDS with GLOBAL_LOAD_ASYNC_TO_LDS_B64 (ASYNCcnt),
// double-buffered: chunk c+1 in flight while recurrence consumes chunk c.
// ---------------------------------------------------------------------------
__global__ __launch_bounds__(256)
void s4_scan(const float* __restrict__ u, const float* __restrict__ A_log,
             const float* __restrict__ Bm, const float* __restrict__ Cm,
             const float* __restrict__ Dp, const float* __restrict__ dt,
             float* __restrict__ yT) {
  __shared__ float ubuf[2][8][64];
  __shared__ float ybuf[8][64];
  const int lane = threadIdx.x & 31;
  const int wave = threadIdx.x >> 5;
  const int g = blockIdx.x * 8 + wave;   // 0..2047
  const int b = g >> 10;
  const int d = g & 1023;

  const float dtd = dt[d];
  const int n0 = lane, n1 = lane + 32;
  const float dA0 = __expf(-__expf(-A_log[d * N_STATE + n0]) * dtd);
  const float dA1 = __expf(-__expf(-A_log[d * N_STATE + n1]) * dtd);
  const float w0  = Cm[d * N_STATE + n0] * Bm[d * N_STATE + n0] * dtd;
  const float w1  = Cm[d * N_STATE + n1] * Bm[d * N_STATE + n1] * dtd;
  const float skip = Dp[d];

  const float* urow = u  + ((size_t)b * D_MODEL + d) * L_SEQ;
  float*       yrow = yT + ((size_t)b * D_MODEL + d) * L_SEQ;

  // async Global->LDS: each of 32 lanes moves 8 bytes (one 64-float chunk/wave)
  auto issue = [&](int c, int bufsel) {
    unsigned ldsoff = (unsigned)(uintptr_t)(&ubuf[bufsel][wave][lane * 2]);
    const float* gp = urow + c * 64 + lane * 2;
    asm volatile("global_load_async_to_lds_b64 %0, %1, off"
                 :: "v"(ldsoff), "v"(gp) : "memory");
  };

  issue(0, 0);
  const int NC = L_SEQ / 64;
  float h0 = 0.f, h1 = 0.f;
  for (int c = 0; c < NC; ++c) {
    const int cur = c & 1;
    if (c + 1 < NC) {
      issue(c + 1, cur ^ 1);
      asm volatile("s_wait_asynccnt 1" ::: "memory");  // oldest (chunk c) complete
    } else {
      asm volatile("s_wait_asynccnt 0" ::: "memory");
    }
    #pragma unroll 4
    for (int t = 0; t < 64; ++t) {
      float ut = ubuf[cur][wave][t];                   // uniform ds_load (broadcast)
      h0 = __builtin_fmaf(dA0, h0, ut);
      h1 = __builtin_fmaf(dA1, h1, ut);
      float p = __builtin_fmaf(w1, h1, w0 * h0);
      p += __shfl_xor(p, 1, 32);
      p += __shfl_xor(p, 2, 32);
      p += __shfl_xor(p, 4, 32);
      p += __shfl_xor(p, 8, 32);
      p += __shfl_xor(p, 16, 32);
      if (lane == 0) ybuf[wave][t] = __builtin_fmaf(skip, ut, p);
    }
    asm volatile("s_wait_dscnt 0" ::: "memory");       // same-wave LDS RAW fence
    if (lane < 16)
      *reinterpret_cast<f32x4*>(&yrow[c * 64 + lane * 4]) =
          *reinterpret_cast<const f32x4*>(&ybuf[wave][lane * 4]);
  }
}

// ---------------------------------------------------------------------------
extern "C" void kernel_launch(void* const* d_in, const int* in_sizes, int n_in,
                              void* d_out, int out_size, void* d_ws, size_t ws_size,
                              hipStream_t stream) {
  (void)in_sizes; (void)n_in; (void)out_size; (void)ws_size;
  const float* x     = (const float*)d_in[0];
  const float* Win   = (const float*)d_in[1];
  const float* bin   = (const float*)d_in[2];
  const float* A_log = (const float*)d_in[3];
  const float* Bm    = (const float*)d_in[4];
  const float* Cm    = (const float*)d_in[5];
  const float* Dp    = (const float*)d_in[6];
  const float* dt    = (const float*)d_in[7];
  const float* Wout  = (const float*)d_in[8];
  const float* bout  = (const float*)d_in[9];
  float* out = (float*)d_out;

  float* x1 = (float*)d_ws;                              // [8192,1024]
  float* u  = x1 + (size_t)M_TOTAL * D_MODEL;            // [2,1024,4096]
  float* yT = x1;                                        // alias: x1 dead after transpose

  dim3 gGemm(M_TOTAL / 128, D_MODEL / 128);
  gemm_bf16x3<false><<<gGemm, dim3(256), 0, stream>>>(x, Win, bin, x1,
                                                      D_MODEL, D_MODEL);
  transpose_ld<<<dim3(L_SEQ / 32, D_MODEL / 32, BATCH), dim3(32, 8), 0, stream>>>(x1, u);
  s4_scan<<<dim3(2048 / 8), dim3(256), 0, stream>>>(u, A_log, Bm, Cm, Dp, dt, yT);
  gemm_bf16x3<true><<<gGemm, dim3(256), 0, stream>>>(yT, Wout, bout, out,
                                                     D_MODEL, D_MODEL);
}
